// MLPEncoder_17643725652588
// MI455X (gfx1250) — compile-verified, matrix operand order checked by reading, not run on hardware
//
#include <hip/hip_runtime.h>

typedef __attribute__((ext_vector_type(16))) __bf16 v16bf;
typedef __attribute__((ext_vector_type(8)))  __bf16 v8bf;
typedef __attribute__((ext_vector_type(8)))  float  v8f;

#define BM 128
#define BNT 128
#define KT 32
#define LDA 40
#define LDB 40

// ---------------------------------------------------------------------------
// fp32 -> bf16 convert (flat)
// ---------------------------------------------------------------------------
__global__ void cvt_f32_bf16(const float* __restrict__ src, __bf16* __restrict__ dst, int n) {
  int i = blockIdx.x * blockDim.x + threadIdx.x;
  if (i < n) dst[i] = (__bf16)src[i];
}

// fp32 W[K][256] -> bf16 Wt[256][K] (N-major so GEMM B-tiles are contiguous)
__global__ void cvt_w_t(const float* __restrict__ src, __bf16* __restrict__ dst, int K) {
  int i = blockIdx.x * blockDim.x + threadIdx.x;
  if (i >= K * 256) return;
  int k = i >> 8, n = i & 255;
  dst[(size_t)n * K + k] = (__bf16)src[i];
}

// ---------------------------------------------------------------------------
// derive indices from one-hot rel_send / rel_rec
// ---------------------------------------------------------------------------
__global__ void build_idx(const float* __restrict__ rel_send, const float* __restrict__ rel_rec,
                          int* __restrict__ sidx, int* __restrict__ ridx, int E) {
  int e = blockIdx.x * blockDim.x + threadIdx.x;
  if (e >= E) return;
  int s = 0, r = 0;
  for (int n = 0; n < 64; ++n) {
    if (rel_send[e * 64 + n] > 0.5f) s = n;
    if (rel_rec[e * 64 + n] > 0.5f) r = n;
  }
  sidx[e] = s;
  ridx[e] = r;
}

__global__ void build_rec_list(const int* __restrict__ ridx, int* __restrict__ rec_cnt,
                               int* __restrict__ rec_list, int E, int perNode) {
  int e = blockIdx.x * blockDim.x + threadIdx.x;
  if (e >= E) return;
  int r = ridx[e];
  int p = atomicAdd(&rec_cnt[r], 1);
  rec_list[r * perNode + p] = e;
}

// ---------------------------------------------------------------------------
// bf16 WMMA GEMM: out[M,256] = ELU(A[M,K] @ W[K,256] + bias)
// block = 256 threads (8 waves), block tile 128x128, wave tile 32x64
// (2x4 WMMA tiles -> 8 v_wmma per K-step), double-buffered LDS, one barrier
// per K-step. A-source row base pointers hoisted out of the K-loop.
// MODE 0 = plain [M,K]; MODE 1 = cat(send,rec) K=512; MODE 2 = +skip K=768.
// Wt is N-major: Wt[n][k].
// ---------------------------------------------------------------------------
template <int MODE>
__global__ void __launch_bounds__(256)
gemm_bias_elu(const __bf16* __restrict__ A, const __bf16* __restrict__ Wt,
              const float* __restrict__ bias, __bf16* __restrict__ out, int M, int K,
              const __bf16* __restrict__ nodeX, const int* __restrict__ sidx,
              const int* __restrict__ ridx, const __bf16* __restrict__ skip, int E) {
  __shared__ alignas(16) __bf16 sA[2][BM * LDA];
  __shared__ alignas(16) __bf16 sB[2][BNT * LDB];

  const int t = threadIdx.x;
  const int lane = t & 31;
  const int wave = t >> 5;
  const int wm = wave & 3;   // M subtile 0..3 (32 rows each)
  const int wn = wave >> 2;  // N subtile 0..1 (64 cols each)
  const int g = lane >> 4;   // lane half
  const int lr = lane & 15;

  const int rowBase = blockIdx.x * BM;
  const int colBase = blockIdx.y * BNT;

  // --- loop-invariant per-thread chunk descriptors -------------------------
  // A tile 128x32: two 8-elem chunks per thread; chunk = i*256 + t
  int rowA[2], kcA[2];
  const __bf16* aS[2];   // MODE0: row ptr (kc folded); MODE1/2: sender row base
  const __bf16* aR[2];   // receiver row base
  const __bf16* aK[2];   // skip row base
#pragma unroll
  for (int i = 0; i < 2; ++i) {
    int chunk = i * 256 + t;
    rowA[i] = chunk >> 2;       // 0..127
    kcA[i] = (chunk & 3) * 8;   // 0,8,16,24
    int rowG = rowBase + rowA[i];
    if constexpr (MODE == 0) {
      aS[i] = A + (size_t)rowG * K + kcA[i];
      aR[i] = nullptr; aK[i] = nullptr;
    } else {
      int b = rowG / E;
      int e = rowG - b * E;
      aS[i] = nodeX + ((size_t)b * 64 + sidx[e]) * 256;
      aR[i] = nodeX + ((size_t)b * 64 + ridx[e]) * 256;
      aK[i] = (MODE == 2) ? (skip + (size_t)rowG * 256) : nullptr;
    }
  }
  // B tile 128(N) x 32(K): two 8-elem chunks per thread
  int nB[2], kcB[2];
  const __bf16* bP[2];
#pragma unroll
  for (int i = 0; i < 2; ++i) {
    int chunk = i * 256 + t;
    nB[i] = chunk >> 2;          // 0..127
    kcB[i] = (chunk & 3) * 8;
    bP[i] = Wt + (size_t)(colBase + nB[i]) * K + kcB[i];
  }

  auto loadTiles = [&](int buf, int kb) {
#pragma unroll
    for (int i = 0; i < 2; ++i) {
      const __bf16* p;
      if constexpr (MODE == 0) {
        p = aS[i] + kb;
      } else {
        int k = kb + kcA[i];
        if constexpr (MODE == 1) {
          p = (k < 256) ? (aS[i] + k) : (aR[i] + (k - 256));
        } else {
          p = (k < 256) ? (aS[i] + k)
                        : ((k < 512) ? (aR[i] + (k - 256)) : (aK[i] + (k - 512)));
        }
      }
      *(v8bf*)&sA[buf][rowA[i] * LDA + kcA[i]] = *(const v8bf*)p;
    }
#pragma unroll
    for (int i = 0; i < 2; ++i)
      *(v8bf*)&sB[buf][nB[i] * LDB + kcB[i]] = *(const v8bf*)(bP[i] + kb);
  };

  v8f acc[2][4] = {};
  int cur = 0;
  loadTiles(0, 0);

  for (int kb = 0; kb < K; kb += KT) {
    __syncthreads();  // stores for buf `cur` visible; reads of buf `cur^1` done
    if (kb + KT < K) loadTiles(cur ^ 1, kb + KT);

    const __bf16* a_ = sA[cur];
    const __bf16* b_ = sB[cur];
    v16bf afrag[2], bfrag[4];
#pragma unroll
    for (int mi = 0; mi < 2; ++mi) {
      int r = wm * 32 + mi * 16 + lr;
      v8bf lo = *(const v8bf*)&a_[r * LDA + 8 * g];
      v8bf hi = *(const v8bf*)&a_[r * LDA + 16 + 8 * g];
#pragma unroll
      for (int i = 0; i < 8; ++i) { afrag[mi][i] = lo[i]; afrag[mi][8 + i] = hi[i]; }
    }
#pragma unroll
    for (int ni = 0; ni < 4; ++ni) {
      int c = wn * 64 + ni * 16 + lr;
      v8bf lo = *(const v8bf*)&b_[c * LDB + 8 * g];
      v8bf hi = *(const v8bf*)&b_[c * LDB + 16 + 8 * g];
#pragma unroll
      for (int i = 0; i < 8; ++i) { bfrag[ni][i] = lo[i]; bfrag[ni][8 + i] = hi[i]; }
    }
#pragma unroll
    for (int mi = 0; mi < 2; ++mi)
#pragma unroll
      for (int ni = 0; ni < 4; ++ni)
        acc[mi][ni] = __builtin_amdgcn_wmma_f32_16x16x32_bf16(
            false, afrag[mi], false, bfrag[ni], (short)0, acc[mi][ni], false, false);
    cur ^= 1;
  }

  // epilogue: bias + ELU (fast exp), store bf16
#pragma unroll
  for (int mi = 0; mi < 2; ++mi) {
#pragma unroll
    for (int ni = 0; ni < 4; ++ni) {
      int col = colBase + wn * 64 + ni * 16 + lr;
      float bv = bias[col];
#pragma unroll
      for (int v = 0; v < 8; ++v) {
        int row = rowBase + wm * 32 + mi * 16 + g * 8 + v;
        float x = acc[mi][ni][v] + bv;
        float y = x > 0.f ? x : (__expf(x) - 1.f);
        out[(size_t)row * 256 + col] = (__bf16)y;
      }
    }
  }
}

// ---------------------------------------------------------------------------
// BatchNorm: column sum / sumsq (atomics into stats[0:256]=sum, [256:512]=sumsq)
// ---------------------------------------------------------------------------
__global__ void bn_stats(const __bf16* __restrict__ x, float* __restrict__ stats, int M,
                         int rowsPerBlock) {
  int c = threadIdx.x;  // 256 threads = columns
  int r0 = blockIdx.x * rowsPerBlock;
  int r1 = min(r0 + rowsPerBlock, M);
  float s = 0.f, s2 = 0.f;
  for (int r = r0; r < r1; ++r) {
    float v = (float)x[(size_t)r * 256 + c];
    s += v;
    s2 += v * v;
  }
  atomicAdd(&stats[c], s);
  atomicAdd(&stats[256 + c], s2);
}

__global__ void bn_apply(__bf16* __restrict__ x, const float* __restrict__ stats,
                         const float* __restrict__ gamma, const float* __restrict__ beta, int M) {
  int c = threadIdx.x;
  float inv = 1.0f / (float)M;
  float mean = stats[c] * inv;
  float var = stats[256 + c] * inv - mean * mean;
  float scale = gamma[c] * rsqrtf(var + 1e-5f);
  float shift = beta[c] - mean * scale;
  for (int r = blockIdx.x; r < M; r += gridDim.x) {
    size_t idx = (size_t)r * 256 + c;
    x[idx] = (__bf16)((float)x[idx] * scale + shift);
  }
}

// ---------------------------------------------------------------------------
// edge2node: incoming[b,n,c] = (1/64) * sum over edges received by node n
// ---------------------------------------------------------------------------
__global__ void edge2node(const __bf16* __restrict__ xe, const int* __restrict__ rec_list,
                          __bf16* __restrict__ xn, int E, int perNode) {
  int bn = blockIdx.x;  // b*64 + n
  int b = bn >> 6, n = bn & 63;
  int c = threadIdx.x;
  float s = 0.f;
  for (int j = 0; j < perNode; ++j) {
    int e = rec_list[n * perNode + j];
    s += (float)xe[((size_t)b * E + e) * 256 + c];
  }
  xn[(size_t)bn * 256 + c] = (__bf16)(s * (1.0f / 64.0f));
}

// ---------------------------------------------------------------------------
// final fc: [M,256] bf16 @ [256,2] fp32 + bias -> fp32. One wave per row.
// ---------------------------------------------------------------------------
__global__ void final_fc(const __bf16* __restrict__ y, const float* __restrict__ w,
                         const float* __restrict__ b, float* __restrict__ out, int M) {
  int row = (int)((blockIdx.x * (size_t)blockDim.x + threadIdx.x) >> 5);
  int lane = threadIdx.x & 31;
  if (row >= M) return;
  float s0 = 0.f, s1 = 0.f;
  for (int c = lane; c < 256; c += 32) {
    float v = (float)y[(size_t)row * 256 + c];
    s0 += v * w[c * 2 + 0];
    s1 += v * w[c * 2 + 1];
  }
#pragma unroll
  for (int off = 16; off > 0; off >>= 1) {
    s0 += __shfl_down(s0, off, 32);
    s1 += __shfl_down(s1, off, 32);
  }
  if (lane == 0) {
    out[(size_t)row * 2 + 0] = s0 + b[0];
    out[(size_t)row * 2 + 1] = s1 + b[1];
  }
}

// ---------------------------------------------------------------------------
// host orchestration
// ---------------------------------------------------------------------------
extern "C" void kernel_launch(void* const* d_in, const int* in_sizes, int n_in, void* d_out,
                              int out_size, void* d_ws, size_t ws_size, hipStream_t stream) {
  const int B = 64, N = 64, H = 256;
  const int E = in_sizes[1] / N;  // 4032
  const int MN = B * N;           // 4096
  const int ME = B * E;           // 258048
  const int perNode = E / N;      // 63

  const float* inputs   = (const float*)d_in[0];
  const float* rel_rec  = (const float*)d_in[1];
  const float* rel_send = (const float*)d_in[2];
  // mlp tuples: (W1, b1, W2, b2, gamma, beta)
  const float* m1[6] = {(const float*)d_in[3],  (const float*)d_in[4],  (const float*)d_in[5],
                        (const float*)d_in[6],  (const float*)d_in[7],  (const float*)d_in[8]};
  const float* m2[6] = {(const float*)d_in[9],  (const float*)d_in[10], (const float*)d_in[11],
                        (const float*)d_in[12], (const float*)d_in[13], (const float*)d_in[14]};
  const float* m3[6] = {(const float*)d_in[15], (const float*)d_in[16], (const float*)d_in[17],
                        (const float*)d_in[18], (const float*)d_in[19], (const float*)d_in[20]};
  const float* m4[6] = {(const float*)d_in[21], (const float*)d_in[22], (const float*)d_in[23],
                        (const float*)d_in[24], (const float*)d_in[25], (const float*)d_in[26]};
  const float* fcw = (const float*)d_in[27];
  const float* fcb = (const float*)d_in[28];
  float* out = (float*)d_out;

  // workspace carve-out
  char* cur = (char*)d_ws;
  auto alloc = [&](size_t bytes) -> void* {
    void* p = cur;
    cur += (bytes + 255) & ~(size_t)255;
    return p;
  };
  __bf16* w1a = (__bf16*)alloc((size_t)256 * 256 * 2);
  __bf16* w1b = (__bf16*)alloc((size_t)256 * 256 * 2);
  __bf16* w2a = (__bf16*)alloc((size_t)512 * 256 * 2);
  __bf16* w2b = (__bf16*)alloc((size_t)256 * 256 * 2);
  __bf16* w3a = (__bf16*)alloc((size_t)256 * 256 * 2);
  __bf16* w3b = (__bf16*)alloc((size_t)256 * 256 * 2);
  __bf16* w4a = (__bf16*)alloc((size_t)768 * 256 * 2);
  __bf16* w4b = (__bf16*)alloc((size_t)256 * 256 * 2);
  int* sidx     = (int*)alloc((size_t)E * 4);
  int* ridx     = (int*)alloc((size_t)E * 4);
  int* rec_cnt  = (int*)alloc((size_t)N * 4);
  int* rec_list = (int*)alloc((size_t)N * perNode * 4);
  float* stats  = (float*)alloc(512 * 4);
  __bf16* nodeIn  = (__bf16*)alloc((size_t)MN * H * 2);
  __bf16* nodeH   = (__bf16*)alloc((size_t)MN * H * 2);
  __bf16* nodeX1  = (__bf16*)alloc((size_t)MN * H * 2);
  __bf16* nodeInc = (__bf16*)alloc((size_t)MN * H * 2);
  __bf16* nodeX3  = (__bf16*)alloc((size_t)MN * H * 2);
  __bf16* edgeH    = (__bf16*)alloc((size_t)ME * H * 2);
  __bf16* edgeSkip = (__bf16*)alloc((size_t)ME * H * 2);
  __bf16* edgeY    = (__bf16*)alloc((size_t)ME * H * 2);

  auto cvtT = [&](const float* s, __bf16* d, int K) {
    cvt_w_t<<<(K * 256 + 255) / 256, 256, 0, stream>>>(s, d, K);
  };
  // weights transposed to [256][K] bf16; inputs to bf16
  cvtT(m1[0], w1a, 256); cvtT(m1[2], w1b, 256);
  cvtT(m2[0], w2a, 512); cvtT(m2[2], w2b, 256);
  cvtT(m3[0], w3a, 256); cvtT(m3[2], w3b, 256);
  cvtT(m4[0], w4a, 768); cvtT(m4[2], w4b, 256);
  cvt_f32_bf16<<<(MN * H + 255) / 256, 256, 0, stream>>>(inputs, nodeIn, MN * H);

  // graph indices
  build_idx<<<(E + 255) / 256, 256, 0, stream>>>(rel_send, rel_rec, sidx, ridx, E);
  hipMemsetAsync(rec_cnt, 0, (size_t)N * 4, stream);
  build_rec_list<<<(E + 255) / 256, 256, 0, stream>>>(ridx, rec_cnt, rec_list, E, perNode);

  auto gemm0 = [&](const __bf16* A, const __bf16* Wt, const float* bias, __bf16* o, int M, int K) {
    dim3 grid(M / BM, 256 / BNT);
    gemm_bias_elu<0><<<grid, 256, 0, stream>>>(A, Wt, bias, o, M, K, nullptr, nullptr, nullptr,
                                               nullptr, E);
  };
  auto bn = [&](__bf16* x, const float* gamma, const float* beta, int M, int rpb) {
    hipMemsetAsync(stats, 0, 512 * 4, stream);
    bn_stats<<<(M + rpb - 1) / rpb, 256, 0, stream>>>(x, stats, M, rpb);
    bn_apply<<<2048, 256, 0, stream>>>(x, stats, gamma, beta, M);
  };

  // mlp1 on nodes
  gemm0(nodeIn, w1a, m1[1], nodeH, MN, 256);
  gemm0(nodeH, w1b, m1[3], nodeX1, MN, 256);
  bn(nodeX1, m1[4], m1[5], MN, 64);

  // mlp2 on edges (fused node2edge gather, K=512)
  {
    dim3 grid(ME / BM, 256 / BNT);
    gemm_bias_elu<1><<<grid, 256, 0, stream>>>(nullptr, w2a, m2[1], edgeH, ME, 512, nodeX1, sidx,
                                               ridx, nullptr, E);
  }
  gemm0(edgeH, w2b, m2[3], edgeSkip, ME, 256);
  bn(edgeSkip, m2[4], m2[5], ME, 512);

  // edge2node
  edge2node<<<MN, 256, 0, stream>>>(edgeSkip, rec_list, nodeInc, E, perNode);

  // mlp3 on nodes
  gemm0(nodeInc, w3a, m3[1], nodeH, MN, 256);
  gemm0(nodeH, w3b, m3[3], nodeX3, MN, 256);
  bn(nodeX3, m3[4], m3[5], MN, 64);

  // mlp4 on edges (fused node2edge + skip concat, K=768)
  {
    dim3 grid(ME / BM, 256 / BNT);
    gemm_bias_elu<2><<<grid, 256, 0, stream>>>(nullptr, w4a, m4[1], edgeH, ME, 768, nodeX3, sidx,
                                               ridx, edgeSkip, E);
  }
  gemm0(edgeH, w4b, m4[3], edgeY, ME, 256);
  bn(edgeY, m4[4], m4[5], ME, 512);

  // final projection
  final_fc<<<(ME + 7) / 8, 256, 0, stream>>>(edgeY, fcw, fcb, out, ME);

  (void)n_in; (void)out_size; (void)ws_size;
}
